// RefinedTimmViT_1786706395797
// MI455X (gfx1250) — compile-verified, matrix operand order checked by reading, not run on hardware
//
#include <hip/hip_runtime.h>
#include <hip/hip_bf16.h>
#include <cmath>

// ---------------- types ----------------
typedef __attribute__((ext_vector_type(16))) _Float16 v16h;
typedef __attribute__((ext_vector_type(8)))  _Float16 v8h;
typedef __attribute__((ext_vector_type(8)))  float    v8f;

#define F_BIAS  1
#define F_GELU  2
#define F_RES   4
#define F_OUT16 8
#define F_OUT32 16

// Model constants
#define Bn   32
#define Dm   768
#define Hn   12
#define DHn  64
#define Ln   12
#define MLPn 3072
#define SHn  128
#define NP1  224   // padded token count, warm phase (197 -> 224, mult of 32)
#define NV1  197
#define NP2  96    // padded token count, pruned phase (65 -> 96)
#define NV2  65
#define M1   (Bn*NP1)  // 7168
#define M2   (Bn*NP2)  // 3072
#define MPAT (Bn*196)  // 6272

// ---------------- WMMA fragment load (ISA 16-bit A/B layout) ----------------
// lane&15 selects the tile row; lanes>=16 hold K offsets +8 / +24.
// v[0..7] = K(k0 + hi8 .. +7), v[8..15] = K(k0+16+hi8 .. +7)
__device__ __forceinline__ v16h load_frag16(const _Float16* __restrict__ p) {
  v8h lo = *(const v8h*)(p);
  v8h hi = *(const v8h*)(p + 16);
  v16h r;
#pragma unroll
  for (int i = 0; i < 8; ++i) { r[i] = lo[i]; r[i + 8] = hi[i]; }
  return r;
}

// ---------------- generic batched WMMA GEMM ----------------
// C[M,N] = A[M,K](f16) * Bt[N,K](f16)^T  (+bias)(+residual)(+gelu)
// blockDim = (32,4): each wave computes a (16*MT)x(16*NT) output tile,
// MT*NT wmmas per 32-wide k-step with full A/B fragment register reuse.
template <int MT, int NT>
__global__ void __launch_bounds__(128)
k_gemm(const _Float16* __restrict__ A, const _Float16* __restrict__ Bt,
       const float* __restrict__ bias, const float* __restrict__ R,
       float* __restrict__ Cf, _Float16* __restrict__ Ch,
       int M, int N, int K, int flags,
       long sA, long sB, long sC)
{
  const int lane = threadIdx.x;
  const int m0 = (blockIdx.y * 4 + threadIdx.y) * (16 * MT);
  if (m0 >= M) return;
  const int n0  = blockIdx.x * (16 * NT);
  const long zb = blockIdx.z;
  const int r   = lane & 15;
  const int hi8 = (lane >> 4) << 3;

  const _Float16* ap[MT];
#pragma unroll
  for (int mi = 0; mi < MT; ++mi)
    ap[mi] = A + zb * sA + (long)(m0 + mi * 16 + r) * K + hi8;
  const _Float16* bp[NT];
#pragma unroll
  for (int ni = 0; ni < NT; ++ni)
    bp[ni] = Bt + zb * sB + (long)(n0 + ni * 16 + r) * K + hi8;

  v8f acc[MT][NT];
#pragma unroll
  for (int mi = 0; mi < MT; ++mi)
#pragma unroll
    for (int ni = 0; ni < NT; ++ni) acc[mi][ni] = (v8f){};

  for (int k0 = 0; k0 < K; k0 += 32) {
    v16h a[MT];
#pragma unroll
    for (int mi = 0; mi < MT; ++mi) a[mi] = load_frag16(ap[mi] + k0);
    v16h b[NT];
#pragma unroll
    for (int ni = 0; ni < NT; ++ni) b[ni] = load_frag16(bp[ni] + k0);
#pragma unroll
    for (int mi = 0; mi < MT; ++mi)
#pragma unroll
      for (int ni = 0; ni < NT; ++ni)
        acc[mi][ni] = __builtin_amdgcn_wmma_f32_16x16x32_f16(
            false, a[mi], false, b[ni], (short)0, acc[mi][ni], false, false);
  }

#pragma unroll
  for (int ni = 0; ni < NT; ++ni) {
    const int col = n0 + ni * 16 + r;
    const float bv = (flags & F_BIAS) ? bias[col] : 0.f;
#pragma unroll
    for (int mi = 0; mi < MT; ++mi) {
#pragma unroll
      for (int i = 0; i < 8; ++i) {
        const int row = m0 + mi * 16 + i + hi8;  // C layout: VGPR i -> M=i (+8 lanes>=16)
        const long o  = zb * sC + (long)row * N + col;
        float v = acc[mi][ni][i] + bv;
        if (flags & F_RES)  v += R[o];
        if (flags & F_GELU) v = 0.5f * v * (1.f + erff(v * 0.70710678118f));
        if (flags & F_OUT32) Cf[o] = v;
        if (flags & F_OUT16) Ch[o] = (_Float16)v;
      }
    }
  }
}

// ---------------- weight convert: f32 [K,N] -> f16 [N,K], batched over z ----
__global__ void k_convt(const float* __restrict__ W, _Float16* __restrict__ Wt,
                        int K, int N)
{
  __shared__ float tile[32][33];
  const long z = blockIdx.z;
  const float* Wz = W + z * (long)K * N;
  _Float16* Wtz = Wt + z * (long)K * N;
  const int kb = blockIdx.y * 32, nb = blockIdx.x * 32;
  const int tx = threadIdx.x, ty = threadIdx.y; // 32x8
  for (int i = ty; i < 32; i += 8) {
    int k = kb + i, n = nb + tx;
    tile[i][tx] = (k < K && n < N) ? Wz[(long)k * N + n] : 0.f;
  }
  __syncthreads();
  for (int i = ty; i < 32; i += 8) {
    int n = nb + i, k = kb + tx;
    if (n < N && k < K) Wtz[(long)n * K + k] = (_Float16)tile[tx][i];
  }
}

__global__ void k_conv16(const float* __restrict__ W, _Float16* __restrict__ O, long n) {
  long i = blockIdx.x * (long)blockDim.x + threadIdx.x;
  if (i < n) O[i] = (_Float16)W[i];
}

// ---------------- im2col for patch embed -> f16 [B*196, 768] ----------------
__global__ void k_im2col(const float* __restrict__ x, _Float16* __restrict__ A) {
  long idx = blockIdx.x * (long)blockDim.x + threadIdx.x;
  const long total = (long)MPAT * Dm;
  if (idx >= total) return;
  int ck = idx % Dm; long t = idx / Dm;
  int p = (int)(t % 196); int b = (int)(t / 196);
  int c = ck >> 8; int ij = ck & 255; int i = ij >> 4; int j = ij & 15;
  int py = p / 14, px = p % 14;
  int iy = py * 16 + i, ix = px * 16 + j;
  A[idx] = (_Float16)x[(((long)b * 3 + c) * 224 + iy) * 224 + ix];
}

// ---------------- assemble tokens: cls + patches + pos_embed ----------------
__global__ void k_assemble(const float* __restrict__ tmp, const float* __restrict__ cls,
                           const float* __restrict__ pos, float* __restrict__ x) {
  long idx = blockIdx.x * (long)blockDim.x + threadIdx.x;
  const long total = (long)Bn * NV1 * Dm;
  if (idx >= total) return;
  int c = idx % Dm; long t2 = idx / Dm;
  int t = (int)(t2 % NV1); int b = (int)(t2 / NV1);
  float v = pos[(long)t * Dm + c];
  if (t == 0) v += cls[c];
  else        v += tmp[((long)b * 196 + (t - 1)) * Dm + c];
  x[((long)b * NP1 + t) * Dm + c] = v;
}

// ---------------- LayerNorm (row of 768), f32 in -> f16 out ----------------
__global__ void __launch_bounds__(256)
k_ln(const float* __restrict__ X, const float* __restrict__ g, const float* __restrict__ b,
     _Float16* __restrict__ O, int rows)
{
  int row = blockIdx.x;
  if (row >= rows) return;
  const float* xr = X + (long)row * Dm;
  __shared__ float red[256];
  int t = threadIdx.x;
  float s = 0.f;
  for (int c = t; c < Dm; c += 256) s += xr[c];
  red[t] = s; __syncthreads();
  for (int o = 128; o > 0; o >>= 1) { if (t < o) red[t] += red[t + o]; __syncthreads(); }
  float mean = red[0] * (1.f / Dm); __syncthreads();
  float vs = 0.f;
  for (int c = t; c < Dm; c += 256) { float d = xr[c] - mean; vs += d * d; }
  red[t] = vs; __syncthreads();
  for (int o = 128; o > 0; o >>= 1) { if (t < o) red[t] += red[t + o]; __syncthreads(); }
  float rstd = rsqrtf(red[0] * (1.f / Dm) + 1e-6f);
  for (int c = t; c < Dm; c += 256)
    O[(long)row * Dm + c] = (_Float16)((xr[c] - mean) * rstd * g[c] + b[c]);
}

// ---------------- split QKV; q scaled by DH^-0.5; V stored transposed ------
__global__ void k_split_qkv(const float* __restrict__ qkv, _Float16* __restrict__ q,
                            _Float16* __restrict__ k, _Float16* __restrict__ vt, int NP)
{
  long idx = blockIdx.x * (long)blockDim.x + threadIdx.x;
  const long total = (long)Bn * NP * Dm;
  if (idx >= total) return;
  int d = idx % DHn; long t = idx / DHn;
  int h = (int)(t % Hn); long t2 = t / Hn;
  int tok = (int)(t2 % NP); int b = (int)(t2 / NP);
  const float* row = qkv + ((long)b * NP + tok) * (3 * Dm) + h * DHn + d;
  long o = (((long)b * Hn + h) * NP + tok) * DHn + d;
  q[o] = (_Float16)(row[0] * 0.125f);       // 1/sqrt(64)
  k[o] = (_Float16)(row[Dm]);
  vt[(((long)b * Hn + h) * DHn + d) * NP + tok] = (_Float16)(row[2 * Dm]);
}

// ---------------- softmax rows (masked to NV valid keys) -> f16 ------------
__global__ void __launch_bounds__(128)
k_softmax(const float* __restrict__ S, _Float16* __restrict__ A, int NP, int NV)
{
  long row = blockIdx.x;
  const float* sr = S + row * (long)NP;
  _Float16* ar = A + row * (long)NP;
  __shared__ float red[128];
  int t = threadIdx.x;
  float mx = -1e30f;
  for (int c = t; c < NV; c += 128) mx = fmaxf(mx, sr[c]);
  red[t] = mx; __syncthreads();
  for (int o = 64; o > 0; o >>= 1) { if (t < o) red[t] = fmaxf(red[t], red[t + o]); __syncthreads(); }
  mx = red[0]; __syncthreads();
  float sum = 0.f;
  for (int c = t; c < NV; c += 128) sum += expf(sr[c] - mx);
  red[t] = sum; __syncthreads();
  for (int o = 64; o > 0; o >>= 1) { if (t < o) red[t] += red[t + o]; __syncthreads(); }
  float inv = 1.f / red[0];
  for (int c = t; c < NP; c += 128)
    ar[c] = (_Float16)(c < NV ? expf(sr[c] - mx) * inv : 0.f);
}

// ---------------- merge heads: [B,H,NP,64] f32 -> [B*NP,768] f16 -----------
__global__ void k_merge(const float* __restrict__ O, _Float16* __restrict__ M16, int NP) {
  long idx = blockIdx.x * (long)blockDim.x + threadIdx.x;
  const long total = (long)Bn * NP * Dm;
  if (idx >= total) return;
  int c = idx % Dm; long t = idx / Dm;
  int tok = (int)(t % NP); int b = (int)(t / NP);
  int h = c >> 6, d = c & 63;
  M16[idx] = (_Float16)O[(((long)b * Hn + h) * NP + tok) * DHn + d];
}

// ---------------- selector: second linear (128->1) + CLS/pad masking -------
__global__ void __launch_bounds__(128)
k_selscore(const _Float16* __restrict__ G, const float* __restrict__ w2,
           const float* __restrict__ b2, float* __restrict__ sc)
{
  int row = blockIdx.x;            // b*NP1 + tok
  int t = threadIdx.x;             // 128
  __shared__ float red[128];
  red[t] = (float)G[(long)row * SHn + t] * w2[t];
  __syncthreads();
  for (int o = 64; o > 0; o >>= 1) { if (t < o) red[t] += red[t + o]; __syncthreads(); }
  if (t == 0) {
    int tok = row % NP1;
    float s = red[0] + b2[0];
    if (tok == 0) s = 1e9f;
    else if (tok >= NV1) s = -1e30f;
    sc[row] = s;
  }
}

// ---------------- top-64 per batch (iterative argmax == top_k order) -------
__global__ void __launch_bounds__(64)
k_topk(const float* __restrict__ sc, int* __restrict__ idx)
{
  int b = blockIdx.x;
  __shared__ float s[NP1];
  __shared__ float bv[64];
  __shared__ int   bi[64];
  int t = threadIdx.x;
  for (int c = t; c < NP1; c += 64) s[c] = sc[b * NP1 + c];
  __syncthreads();
  for (int it = 0; it < 64; ++it) {
    float best = -1e38f; int besti = NP1;
    for (int c = 1 + t; c < NV1; c += 64) {    // tokens 1..196
      float v = s[c];
      if (v > best) { best = v; besti = c; }   // increasing c -> first max kept
    }
    bv[t] = best; bi[t] = besti; __syncthreads();
    for (int o = 32; o > 0; o >>= 1) {
      if (t < o) {
        if (bv[t + o] > bv[t] || (bv[t + o] == bv[t] && bi[t + o] < bi[t])) {
          bv[t] = bv[t + o]; bi[t] = bi[t + o];
        }
      }
      __syncthreads();
    }
    if (t == 0) { idx[b * (1 + 64) + 1 + it] = bi[0]; s[bi[0]] = -1e38f; }
    __syncthreads();
  }
  if (t == 0) idx[b * (1 + 64)] = 0;
}

// ---------------- gather kept tokens (+zero pad rows) ----------------------
__global__ void k_gather(const float* __restrict__ X, const int* __restrict__ idx,
                         float* __restrict__ XR)
{
  long i = blockIdx.x * (long)blockDim.x + threadIdx.x;
  const long total = (long)Bn * NP2 * Dm;
  if (i >= total) return;
  int c = i % Dm; long t = i / Dm;
  int j = (int)(t % NP2); int b = (int)(t / NP2);
  float v = 0.f;
  if (j < NV2) { int tok = idx[b * (1 + 64) + j]; v = X[((long)b * NP1 + tok) * Dm + c]; }
  XR[((long)b * NP2 + j) * Dm + c] = v;
}

// ---------------- final LN on CLS + classifier head ------------------------
__global__ void __launch_bounds__(256)
k_head(const float* __restrict__ XR, const float* __restrict__ g, const float* __restrict__ bn,
       const float* __restrict__ hw, const float* __restrict__ hb, float* __restrict__ out)
{
  int b = blockIdx.x;
  __shared__ float x[Dm];
  __shared__ float red[256];
  int t = threadIdx.x;
  const float* xr = XR + (long)b * NP2 * Dm;   // CLS row
  for (int c = t; c < Dm; c += 256) x[c] = xr[c];
  __syncthreads();
  float s = 0.f;
  for (int c = t; c < Dm; c += 256) s += x[c];
  red[t] = s; __syncthreads();
  for (int o = 128; o > 0; o >>= 1) { if (t < o) red[t] += red[t + o]; __syncthreads(); }
  float mean = red[0] * (1.f / Dm); __syncthreads();
  float vs = 0.f;
  for (int c = t; c < Dm; c += 256) { float d = x[c] - mean; vs += d * d; }
  red[t] = vs; __syncthreads();
  for (int o = 128; o > 0; o >>= 1) { if (t < o) red[t] += red[t + o]; __syncthreads(); }
  float rstd = rsqrtf(red[0] * (1.f / Dm) + 1e-6f); __syncthreads();
  for (int c = t; c < Dm; c += 256) x[c] = (x[c] - mean) * rstd * g[c] + bn[c];
  __syncthreads();
  if (t < 10) {
    float acc = hb[t];
    for (int c = 0; c < Dm; ++c) acc += x[c] * hw[(long)c * 10 + t];
    out[b * 10 + t] = acc;
  }
}

// ============================ host launch ==================================
extern "C" void kernel_launch(void* const* d_in, const int* in_sizes, int n_in,
                              void* d_out, int out_size, void* d_ws, size_t ws_size,
                              hipStream_t stream)
{
  (void)in_sizes; (void)n_in; (void)out_size; (void)ws_size;

  const float* in_x      = (const float*)d_in[0];
  const float* patch_w   = (const float*)d_in[1];
  const float* patch_b   = (const float*)d_in[2];
  const float* cls_token = (const float*)d_in[3];
  const float* pos_embed = (const float*)d_in[4];
  const float* ln1_g = (const float*)d_in[5];
  const float* ln1_b = (const float*)d_in[6];
  const float* qkv_w = (const float*)d_in[7];
  const float* qkv_b = (const float*)d_in[8];
  const float* proj_w = (const float*)d_in[9];
  const float* proj_b = (const float*)d_in[10];
  const float* ln2_g = (const float*)d_in[11];
  const float* ln2_b = (const float*)d_in[12];
  const float* fc1_w = (const float*)d_in[13];
  const float* fc1_b = (const float*)d_in[14];
  const float* fc2_w = (const float*)d_in[15];
  const float* fc2_b = (const float*)d_in[16];
  const float* sel_ln_g = (const float*)d_in[17];
  const float* sel_ln_b = (const float*)d_in[18];
  const float* sel_w1 = (const float*)d_in[19];
  const float* sel_b1 = (const float*)d_in[20];
  const float* sel_w2 = (const float*)d_in[21];
  const float* sel_b2 = (const float*)d_in[22];
  const float* norm_g = (const float*)d_in[23];
  const float* norm_b = (const float*)d_in[24];
  const float* head_w = (const float*)d_in[25];
  const float* head_b = (const float*)d_in[26];
  float* out = (float*)d_out;

  // ---- workspace carve-out ----
  size_t off = 0;
  auto alloc = [&](size_t bytes) -> void* {
    void* p = (void*)((char*)d_ws + off);
    off = (off + bytes + 255) & ~(size_t)255;
    return p;
  };
  _Float16* qkvWt  = (_Float16*)alloc((size_t)Ln * 3 * Dm * Dm * 2);
  _Float16* projWt = (_Float16*)alloc((size_t)Ln * Dm * Dm * 2);
  _Float16* fc1Wt  = (_Float16*)alloc((size_t)Ln * MLPn * Dm * 2);
  _Float16* fc2Wt  = (_Float16*)alloc((size_t)Ln * Dm * MLPn * 2);
  _Float16* patWt  = (_Float16*)alloc((size_t)Dm * Dm * 2);
  _Float16* selW1t = (_Float16*)alloc((size_t)SHn * Dm * 2);

  float*    x1    = (float*)   alloc((size_t)M1 * Dm * 4);
  float*    xB    = (float*)   alloc((size_t)M2 * Dm * 4);
  _Float16* h16   = (_Float16*)alloc((size_t)M1 * Dm * 2);          // LN out / im2col A
  float*    qkvf  = (float*)   alloc((size_t)M1 * 3 * Dm * 4);      // also patch-GEMM tmp
  _Float16* q16   = (_Float16*)alloc((size_t)Bn * Hn * NP1 * DHn * 2);
  _Float16* k16   = (_Float16*)alloc((size_t)Bn * Hn * NP1 * DHn * 2);
  _Float16* vt16  = (_Float16*)alloc((size_t)Bn * Hn * DHn * NP1 * 2);
  float*    scf   = (float*)   alloc((size_t)Bn * Hn * NP1 * NP1 * 4);
  _Float16* at16  = (_Float16*)alloc((size_t)Bn * Hn * NP1 * NP1 * 2);
  float*    of    = (float*)   alloc((size_t)Bn * Hn * NP1 * DHn * 4);
  _Float16* om16  = (_Float16*)alloc((size_t)M1 * Dm * 2);          // also selector-hidden f16
  _Float16* g16   = (_Float16*)alloc((size_t)M1 * MLPn * 2);
  float*    selsc = (float*)   alloc((size_t)Bn * NP1 * 4);
  int*      selix = (int*)     alloc((size_t)Bn * (1 + 64) * 4);

  auto gemm = [&](const _Float16* A, const _Float16* Bt, const float* bias, const float* R,
                  float* Cf, _Float16* Ch, int M, int N, int K, int flags,
                  int batch, long sA, long sB, long sC) {
    dim3 bk(32, 4);
    if (N % 64 == 0) {
      // 32x64 wave tile: 8 wmmas per k-step, best fragment reuse
      dim3 gd((unsigned)(N / 64), (unsigned)((M + 127) / 128), (unsigned)batch);
      k_gemm<2, 4><<<gd, bk, 0, stream>>>(A, Bt, bias, R, Cf, Ch, M, N, K, flags, sA, sB, sC);
    } else {
      // fallback 16x32 tile (score GEMM: N = 224 / 96)
      dim3 gd((unsigned)(N / 32), (unsigned)((M + 63) / 64), (unsigned)batch);
      k_gemm<1, 2><<<gd, bk, 0, stream>>>(A, Bt, bias, R, Cf, Ch, M, N, K, flags, sA, sB, sC);
    }
  };

  // ---- weight conversion (fp32 -> f16, transposed to [N][K]) ----
  {
    dim3 bk(32, 8);
    k_convt<<<dim3(3 * Dm / 32, Dm / 32, Ln), bk, 0, stream>>>(qkv_w, qkvWt, Dm, 3 * Dm);
    k_convt<<<dim3(Dm / 32, Dm / 32, Ln),     bk, 0, stream>>>(proj_w, projWt, Dm, Dm);
    k_convt<<<dim3(MLPn / 32, Dm / 32, Ln),   bk, 0, stream>>>(fc1_w, fc1Wt, Dm, MLPn);
    k_convt<<<dim3(Dm / 32, MLPn / 32, Ln),   bk, 0, stream>>>(fc2_w, fc2Wt, MLPn, Dm);
    k_convt<<<dim3(SHn / 32, Dm / 32, 1),     bk, 0, stream>>>(sel_w1, selW1t, Dm, SHn);
    long npw = (long)Dm * Dm;   // patch_w is already [out=768][cij=768] == Bt layout
    k_conv16<<<(unsigned)((npw + 255) / 256), 256, 0, stream>>>(patch_w, patWt, npw);
  }

  // ---- patch embed: im2col + GEMM, assemble tokens ----
  {
    long n = (long)MPAT * Dm;
    k_im2col<<<(unsigned)((n + 255) / 256), 256, 0, stream>>>(in_x, h16);
    gemm(h16, patWt, patch_b, nullptr, qkvf, nullptr, MPAT, Dm, Dm, F_BIAS | F_OUT32, 1, 0, 0, 0);
    hipMemsetAsync((void*)x1, 0, (size_t)M1 * Dm * 4, stream);
    long na = (long)Bn * NV1 * Dm;
    k_assemble<<<(unsigned)((na + 255) / 256), 256, 0, stream>>>(qkvf, cls_token, pos_embed, x1);
  }

  // ---- transformer blocks ----
  for (int i = 0; i < Ln; ++i) {
    const bool warm = (i < 2);
    const int NP = warm ? NP1 : NP2;
    const int NV = warm ? NV1 : NV2;
    float* X = warm ? x1 : xB;
    const int M = Bn * NP;
    const int BH = Bn * Hn;
    const long sqk = (long)NP * DHn;
    const long ssc = (long)NP * NP;

    // attention
    k_ln<<<M, 256, 0, stream>>>(X, ln1_g + (long)i * Dm, ln1_b + (long)i * Dm, h16, M);
    gemm(h16, qkvWt + (long)i * 3 * Dm * Dm, qkv_b + (long)i * 3 * Dm, nullptr,
         qkvf, nullptr, M, 3 * Dm, Dm, F_BIAS | F_OUT32, 1, 0, 0, 0);
    {
      long n = (long)M * Dm;
      k_split_qkv<<<(unsigned)((n + 255) / 256), 256, 0, stream>>>(qkvf, q16, k16, vt16, NP);
    }
    gemm(q16, k16, nullptr, nullptr, scf, nullptr, NP, NP, DHn, F_OUT32, BH, sqk, sqk, ssc);
    k_softmax<<<(unsigned)(BH * NP), 128, 0, stream>>>(scf, at16, NP, NV);
    gemm(at16, vt16, nullptr, nullptr, of, nullptr, NP, DHn, NP, F_OUT32, BH, ssc, sqk, sqk);
    {
      long n = (long)M * Dm;
      k_merge<<<(unsigned)((n + 255) / 256), 256, 0, stream>>>(of, om16, NP);
    }
    gemm(om16, projWt + (long)i * Dm * Dm, proj_b + (long)i * Dm, X,
         X, nullptr, M, Dm, Dm, F_BIAS | F_RES | F_OUT32, 1, 0, 0, 0);

    // MLP
    k_ln<<<M, 256, 0, stream>>>(X, ln2_g + (long)i * Dm, ln2_b + (long)i * Dm, h16, M);
    gemm(h16, fc1Wt + (long)i * MLPn * Dm, fc1_b + (long)i * MLPn, nullptr,
         nullptr, g16, M, MLPn, Dm, F_BIAS | F_GELU | F_OUT16, 1, 0, 0, 0);
    gemm(g16, fc2Wt + (long)i * Dm * MLPn, fc2_b + (long)i * Dm, X,
         X, nullptr, M, Dm, MLPn, F_BIAS | F_RES | F_OUT32, 1, 0, 0, 0);

    // token selection after the 2 warm blocks
    if (i == 1) {
      k_ln<<<M1, 256, 0, stream>>>(x1, sel_ln_g, sel_ln_b, h16, M1);
      gemm(h16, selW1t, sel_b1, nullptr, nullptr, om16, M1, SHn, Dm,
           F_BIAS | F_GELU | F_OUT16, 1, 0, 0, 0);
      k_selscore<<<M1, 128, 0, stream>>>(om16, sel_w2, sel_b2, selsc);
      k_topk<<<Bn, 64, 0, stream>>>(selsc, selix);
      long n = (long)M2 * Dm;
      k_gather<<<(unsigned)((n + 255) / 256), 256, 0, stream>>>(x1, selix, xB);
    }
  }

  // ---- final LN on CLS + head ----
  k_head<<<Bn, 256, 0, stream>>>(xB, norm_g, norm_b, head_w, head_b, out);
}
